// SiameseGPSite_49512382988755
// MI455X (gfx1250) — matrix operand order repaired and checked by reading, not compiled, not run on hardware
//
#include <hip/hip_runtime.h>
#include <hip/hip_bf16.h>
#include <math.h>

// ---------------------------------------------------------------------------
// Types for CDNA5 WMMA (wave32, 16x16x32 bf16 -> f32)
// ---------------------------------------------------------------------------
typedef __bf16 bf16;
typedef __attribute__((ext_vector_type(16))) __bf16 v16bf;
typedef __attribute__((ext_vector_type(8)))  __bf16 v8bf;
typedef __attribute__((ext_vector_type(4)))  __bf16 v4bf;
typedef __attribute__((ext_vector_type(8)))  float  v8f;

union Frag { v16bf v; v8bf h[2]; };

// ---------------------------------------------------------------------------
// small device helpers
// ---------------------------------------------------------------------------
__device__ __forceinline__ void sub3(const float a[3], const float b[3], float o[3]) {
    o[0] = a[0] - b[0]; o[1] = a[1] - b[1]; o[2] = a[2] - b[2];
}
__device__ __forceinline__ float dot3(const float a[3], const float b[3]) {
    return a[0]*b[0] + a[1]*b[1] + a[2]*b[2];
}
__device__ __forceinline__ void cross3(const float a[3], const float b[3], float o[3]) {
    o[0] = a[1]*b[2] - a[2]*b[1];
    o[1] = a[2]*b[0] - a[0]*b[2];
    o[2] = a[0]*b[1] - a[1]*b[0];
}
__device__ __forceinline__ void norm3(float v[3]) {
    float l = sqrtf(dot3(v, v));
    float s = 1.0f / fmaxf(l, 1e-12f);
    v[0] *= s; v[1] *= s; v[2] *= s;
}
__device__ __forceinline__ float clampf(float x, float lo, float hi) {
    return fminf(fmaxf(x, lo), hi);
}
__device__ __forceinline__ float sgnf(float x) {
    return (x > 0.f) ? 1.f : ((x < 0.f) ? -1.f : 0.f);
}
// monotone float <-> signed-int encoding for atomicMax-based segment max
__device__ __forceinline__ int fenc(float f) {
    int i = __float_as_int(f);
    return (i >= 0) ? i : (i ^ 0x7fffffff);
}
__device__ __forceinline__ float fdec(int k) {
    int i = (k >= 0) ? k : (k ^ 0x7fffffff);
    return __int_as_float(i);
}
#define MKEY_SENTINEL ((int)0x80000000)

#define XIN_LDA 1248   // 1217 padded to 32
#define EIN_LDA 480    // 450 padded to 32

// ---------------------------------------------------------------------------
// fills / packing
// ---------------------------------------------------------------------------
__global__ void gps_fill_f32(float* p, float v, long long n) {
    long long i = (long long)blockIdx.x * 256 + threadIdx.x;
    if (i < n) p[i] = v;
}
__global__ void gps_fill_i32(int* p, int v, long long n) {
    long long i = (long long)blockIdx.x * 256 + threadIdx.x;
    if (i < n) p[i] = v;
}
// f32 (M,K) row-major -> bf16 (M,Kpad) zero-padded
__global__ void gps_pack_w(const float* __restrict__ W, bf16* __restrict__ Wp,
                           int M, int K, int Kpad) {
    long long i = (long long)blockIdx.x * 256 + threadIdx.x;
    if (i >= (long long)M * Kpad) return;
    int r = (int)(i / Kpad), c = (int)(i - (long long)r * Kpad);
    float v = (c < K) ? W[(size_t)r * K + c] : 0.f;
    Wp[i] = (bf16)v;
}
// zero columns [K, Kpad) of a (rows, Kpad) f32 buffer
__global__ void gps_pad_zero(float* __restrict__ buf, long long rows, int K, int Kpad) {
    int padw = Kpad - K;
    long long i = (long long)blockIdx.x * 256 + threadIdx.x;
    if (i >= rows * padw) return;
    long long r = i / padw;
    int c = K + (int)(i - r * padw);
    buf[r * (long long)Kpad + c] = 0.f;
}

// ---------------------------------------------------------------------------
// GEMM: Y(N,M) = act( X(N,kpad f32) @ Wp(M,kpad bf16)^T + bias )
// X row stride = kpad (multiple of 32, 128B-aligned rows); W pre-packed bf16.
// block: 256 threads (8 waves) -> 128 rows x 64 cols; wave = 16 rows x 64 cols
// act: 0 none, 1 relu, 2 gelu(exact).  M must be a multiple of 64.
// ---------------------------------------------------------------------------
__global__ __launch_bounds__(256) void gps_gemm(const float* __restrict__ X,
                                                const bf16* __restrict__ Wp,
                                                const float* __restrict__ bias,
                                                float* __restrict__ Y,
                                                int N, int kpad, int M, int act)
{
    __shared__ __align__(16) bf16 sX[128 * 32];
    __shared__ __align__(16) bf16 sW[64 * 32];
    const int tid  = threadIdx.x;
    const int wave = tid >> 5;
    const int lane = tid & 31;
    const int row0 = blockIdx.x * 128;
    const int col0 = blockIdx.y * 64;
    const int half = lane >> 4;
    const int lr   = lane & 15;

    v8f acc[4] = {};

    for (int k0 = 0; k0 < kpad; k0 += 32) {
        // stage X tile (128 rows x 32 K) as bf16: vectorized float4 -> 4x bf16
        #pragma unroll
        for (int it = 0; it < 4; ++it) {
            int i  = tid + it * 256;       // 0..1023
            int r  = i >> 3;               // row in tile
            int ch = i & 7;                // float4 chunk
            int gr = row0 + r;
            if (gr > N - 1) gr = N - 1;    // clamp: unconditional aligned load
            const float* xp = X + (size_t)gr * kpad + k0 + ch * 4;
            float4 f = *(const float4*)xp;
            // speculative WGP-scope prefetch of the next k-slice (harmless OOB)
            __builtin_prefetch((const void*)(xp + 32), 0, 3);
            v4bf pk;
            pk[0] = (bf16)f.x; pk[1] = (bf16)f.y; pk[2] = (bf16)f.z; pk[3] = (bf16)f.w;
            *(v4bf*)(&sX[r * 32 + ch * 4]) = pk;
        }
        // stage W tile (64 rows x 32 K): straight 16B bf16 copy, no guards
        {
            int r  = tid >> 2;             // 0..63
            int ch = tid & 3;              // 8-bf16 chunk
            const v8bf wv = *(const v8bf*)(Wp + (size_t)(col0 + r) * kpad + k0 + ch * 8);
            *(v8bf*)(&sW[r * 32 + ch * 8]) = wv;
        }
        __syncthreads();

        // A fragment: 16 rows of this wave; lanes 0-15: K {0..7,16..23}, lanes 16-31: K {8..15,24..31}
        Frag a;
        const bf16* ap = &sX[(wave * 16 + lr) * 32 + half * 8];
        a.h[0] = *(const v8bf*)(ap);
        a.h[1] = *(const v8bf*)(ap + 16);

        #pragma unroll
        for (int ct = 0; ct < 4; ++ct) {
            Frag b;
            b.h[0] = *(const v8bf*)(&sW[(ct * 16 + lr) * 32 + half * 8]);
            b.h[1] = *(const v8bf*)(&sW[(ct * 16 + lr) * 32 + half * 8 + 16]);
            acc[ct] = __builtin_amdgcn_wmma_f32_16x16x32_bf16(
                false, a.v, false, b.v, (short)0, acc[ct], false, false);
        }
        __syncthreads();
    }

    // epilogue: D element (m = v + 8*half, n = lane&15)
    #pragma unroll
    for (int ct = 0; ct < 4; ++ct) {
        int col = col0 + ct * 16 + lr;
        float bv = bias[col];
        #pragma unroll
        for (int v = 0; v < 8; ++v) {
            int rr = row0 + wave * 16 + v + 8 * half;
            if (rr >= N) continue;
            float val = acc[ct][v] + bv;
            if (act == 1)      val = fmaxf(val, 0.f);
            else if (act == 2) val = 0.5f * val * (1.f + erff(val * 0.70710678118654752f));
            Y[(size_t)rr * M + col] = val;
        }
    }
}

// ---------------------------------------------------------------------------
// geometry: per-node features (184) at xin[n*XIN_LDA+1033..], plus local coords
// ---------------------------------------------------------------------------
__device__ __forceinline__ void loadAtom(const float* X, int node, int atom, float o[3]) {
    const float* p = X + ((size_t)node * 5 + atom) * 3;
    o[0] = p[0]; o[1] = p[1]; o[2] = p[2];
}
__device__ __forceinline__ void getU(const float* X, int t, float u[3]) {
    int n0 = t / 3,  a0 = t - n0 * 3;
    int n1 = (t + 1) / 3, a1 = (t + 1) - n1 * 3;
    float p0[3], p1[3];
    loadAtom(X, n0, a0, p0);
    loadAtom(X, n1, a1, p1);
    sub3(p1, p0, u);
    norm3(u);
}

__global__ __launch_bounds__(128) void gps_geo_node(const float* __restrict__ X,
                                                    float* __restrict__ xin,
                                                    float* __restrict__ lcg, int N)
{
    int n = blockIdx.x * 128 + threadIdx.x;
    if (n >= N) return;
    float a[5][3];
    #pragma unroll
    for (int i = 0; i < 5; i++) loadAtom(X, n, i, a[i]);

    float u[3], v[3], bb[3], nn[3], c2[3];
    sub3(a[1], a[0], u); norm3(u);
    sub3(a[2], a[1], v); norm3(v);
    sub3(u, v, bb); norm3(bb);
    cross3(u, v, nn); norm3(nn);
    cross3(bb, nn, c2);
    float* L = lcg + (size_t)n * 9;
    #pragma unroll
    for (int d = 0; d < 3; d++) { L[d*3+0] = bb[d]; L[d*3+1] = nn[d]; L[d*3+2] = c2[d]; }

    float* o = xin + (size_t)n * XIN_LDA + 1033;

    // _node_angle on the flattened 3N backbone chain
    #pragma unroll
    for (int j = 0; j < 3; j++) {
        int s = 3 * n + j - 1;
        float D = 0.f, B = 0.f;
        if (s >= 0 && s <= 3 * N - 4) {
            float u2[3], u1[3], u0[3];
            getU(X, s, u2); getU(X, s + 1, u1); getU(X, s + 2, u0);
            float n2[3], n1[3];
            cross3(u2, u1, n2); norm3(n2);
            cross3(u1, u0, n1); norm3(n1);
            float cD = clampf(dot3(n2, n1), -1.f + 1e-7f, 1.f - 1e-7f);
            D = sgnf(dot3(u2, n1)) * acosf(cD);
            float cB = clampf(dot3(u2, u1), -1.f + 1e-7f, 1.f - 1e-7f);
            B = acosf(cB);
        }
        o[j]     = cosf(D);
        o[3 + j] = sinf(D);
        o[6 + j] = cosf(B);
        o[9 + j] = sinf(B);
    }

    // node_dist: 10 atom-pairs x 16 RBF
    const int PA[10] = {1, 1, 1, 0, 0, 3, 4, 4, 4, 4};
    const int PB[10] = {0, 2, 3, 2, 3, 2, 0, 1, 2, 3};
    #pragma unroll
    for (int p = 0; p < 10; p++) {
        float t[3];
        sub3(a[PA[p]], a[PB[p]], t);
        float d = sqrtf(dot3(t, t));
        #pragma unroll
        for (int k = 0; k < 16; k++) {
            float mu = (20.f / 15.f) * (float)k;
            float z = (d - mu) * 0.8f;
            o[12 + p * 16 + k] = expf(-z * z);
        }
    }

    // node_dir: 4 unit directions projected onto the local frame
    const int TI[4] = {0, 2, 3, 4};
    #pragma unroll
    for (int t = 0; t < 4; t++) {
        float tv[3];
        sub3(a[TI[t]], a[1], tv); norm3(tv);
        #pragma unroll
        for (int col = 0; col < 3; col++) {
            o[172 + t * 3 + col] = tv[0]*L[0*3+col] + tv[1]*L[1*3+col] + tv[2]*L[2*3+col];
        }
    }
}

// ---------------------------------------------------------------------------
// per-edge 450 geometric features (row stride EIN_LDA)
// ---------------------------------------------------------------------------
__global__ __launch_bounds__(128) void gps_geo_edge(const float* __restrict__ X,
                                                    const float* __restrict__ lcg,
                                                    const int* __restrict__ EI,
                                                    float* __restrict__ ein, int E)
{
    int e = blockIdx.x * 128 + threadIdx.x;
    if (e >= E) return;
    int src = EI[e], dst = EI[E + e];

    float Xs[5][3], Xd[5][3], Ls[9], Ld[9];
    #pragma unroll
    for (int i = 0; i < 5; i++) { loadAtom(X, src, i, Xs[i]); loadAtom(X, dst, i, Xd[i]); }
    #pragma unroll
    for (int i = 0; i < 9; i++) { Ls[i] = lcg[(size_t)src * 9 + i]; Ld[i] = lcg[(size_t)dst * 9 + i]; }

    float* o = ein + (size_t)e * EIN_LDA;

    // positional encoding (16)
    float dd = (float)(src - dst);
    #pragma unroll
    for (int k = 0; k < 8; k++) {
        float fr = expf(-(float)(2 * k) * 0.57564627324851f); // ln(1e4)/16
        float ang = dd * fr;
        o[k] = cosf(ang);
        o[8 + k] = sinf(ang);
    }

    // orientation quaternion (4): R = lc[dst]^T * lc[src]
    float R[3][3];
    #pragma unroll
    for (int ai = 0; ai < 3; ai++)
        #pragma unroll
        for (int bi = 0; bi < 3; bi++)
            R[ai][bi] = Ld[0*3+ai]*Ls[0*3+bi] + Ld[1*3+ai]*Ls[1*3+bi] + Ld[2*3+ai]*Ls[2*3+bi];
    float m0 = 0.5f * sqrtf(fabsf(1.f + R[0][0] - R[1][1] - R[2][2]));
    float m1 = 0.5f * sqrtf(fabsf(1.f - R[0][0] + R[1][1] - R[2][2]));
    float m2 = 0.5f * sqrtf(fabsf(1.f - R[0][0] - R[1][1] + R[2][2]));
    float q0 = sgnf(R[2][1] - R[1][2]) * m0;
    float q1 = sgnf(R[0][2] - R[2][0]) * m1;
    float q2 = sgnf(R[1][0] - R[0][1]) * m2;
    float qw = 0.5f * sqrtf(fmaxf(0.f, 1.f + R[0][0] + R[1][1] + R[2][2]));
    float ql = fmaxf(sqrtf(q0*q0 + q1*q1 + q2*q2 + qw*qw), 1e-12f);
    o[16] = q0 / ql; o[17] = q1 / ql; o[18] = q2 / ql; o[19] = qw / ql;

    // edge_dist: 25 atom-pairs x 16 RBF (400)
    #pragma unroll
    for (int ai = 0; ai < 5; ai++) {
        #pragma unroll
        for (int bi = 0; bi < 5; bi++) {
            float t[3];
            sub3(Xs[ai], Xd[bi], t);
            float d = sqrtf(dot3(t, t));
            float* od = o + 20 + (ai * 5 + bi) * 16;
            #pragma unroll
            for (int k = 0; k < 16; k++) {
                float mu = (20.f / 15.f) * (float)k;
                float z = (d - mu) * 0.8f;
                od[k] = expf(-z * z);
            }
        }
    }

    // edge_dir (30)
    #pragma unroll
    for (int ai = 0; ai < 5; ai++) {
        float t[3];
        sub3(Xs[ai], Xd[1], t); norm3(t);
        #pragma unroll
        for (int col = 0; col < 3; col++)
            o[420 + ai * 3 + col] = t[0]*Ld[0*3+col] + t[1]*Ld[1*3+col] + t[2]*Ld[2*3+col];
    }
    #pragma unroll
    for (int ai = 0; ai < 5; ai++) {
        float t[3];
        sub3(Xd[ai], Xs[1], t); norm3(t);
        #pragma unroll
        for (int col = 0; col < 3; col++)
            o[435 + ai * 3 + col] = t[0]*Ls[0*3+col] + t[1]*Ls[1*3+col] + t[2]*Ls[2*3+col];
    }
}

// ---------------------------------------------------------------------------
// copy h_V into the XIN_LDA-wide node input buffer
// ---------------------------------------------------------------------------
__global__ __launch_bounds__(256) void gps_copy_hv(const float* __restrict__ hv,
                                                   float* __restrict__ xin, int N)
{
    int n = blockIdx.x;
    if (n >= N) return;
    for (int j = threadIdx.x; j < 1033; j += 256)
        xin[(size_t)n * XIN_LDA + j] = hv[(size_t)n * 1033 + j];
}

// ---------------------------------------------------------------------------
// batch norm (feature dim = 128): stats then apply (optionally summing A+B)
// ---------------------------------------------------------------------------
__global__ __launch_bounds__(256) void gps_bn_stats(const float* __restrict__ A,
                                                    const float* __restrict__ B,
                                                    float* __restrict__ stats,
                                                    long long total)
{
    __shared__ float ss[128], sq[128];
    int tid = threadIdx.x;
    if (tid < 128) { ss[tid] = 0.f; sq[tid] = 0.f; }
    __syncthreads();
    for (long long i = (long long)blockIdx.x * 256 + tid; i < total;
         i += (long long)gridDim.x * 256) {
        float x = A[i];
        if (B) x += B[i];
        int c = (int)(i & 127);
        atomicAdd(&ss[c], x);
        atomicAdd(&sq[c], x * x);
    }
    __syncthreads();
    if (tid < 128) {
        atomicAdd(&stats[tid], ss[tid]);
        atomicAdd(&stats[128 + tid], sq[tid]);
    }
}

__global__ __launch_bounds__(256) void gps_bn_apply(const float* __restrict__ A,
                                                    const float* __restrict__ B,
                                                    const float* __restrict__ stats,
                                                    const float* __restrict__ g,
                                                    const float* __restrict__ b,
                                                    float* __restrict__ Y,
                                                    long long total, float invN)
{
    long long i = (long long)blockIdx.x * 256 + threadIdx.x;
    if (i >= total) return;
    int c = (int)(i & 127);
    float mean = stats[c] * invN;
    float var  = stats[128 + c] * invN - mean * mean;
    float x = A[i];
    if (B) x += B[i];
    Y[i] = g[c] * (x - mean) * rsqrtf(var + 1e-5f) + b[c];
}

// ---------------------------------------------------------------------------
// layer norm over 128 features with residual add: Y = LN(A + B); wave per row
// ---------------------------------------------------------------------------
__global__ __launch_bounds__(256) void gps_ln(const float* __restrict__ A,
                                              const float* __restrict__ B,
                                              const float* __restrict__ g,
                                              const float* __restrict__ b,
                                              float* __restrict__ Y, int N)
{
    int wave = threadIdx.x >> 5, lane = threadIdx.x & 31;
    int row = blockIdx.x * 8 + wave;
    if (row >= N) return;
    const float* ar = A + (size_t)row * 128;
    const float* br = B + (size_t)row * 128;
    float v[4], s = 0.f, s2 = 0.f;
    #pragma unroll
    for (int i = 0; i < 4; i++) {
        int c = lane * 4 + i;
        float x = ar[c] + br[c];
        v[i] = x; s += x; s2 += x * x;
    }
    #pragma unroll
    for (int o = 16; o > 0; o >>= 1) { s += __shfl_xor(s, o); s2 += __shfl_xor(s2, o); }
    float mean = s * (1.f / 128.f);
    float var  = s2 * (1.f / 128.f) - mean * mean;
    float r = rsqrtf(var + 1e-5f);
    #pragma unroll
    for (int i = 0; i < 4; i++) {
        int c = lane * 4 + i;
        Y[(size_t)row * 128 + c] = g[c] * (v[i] - mean) * r + b[c];
    }
}

// ---------------------------------------------------------------------------
// attention: alpha + segment max (wave per (edge,head))
// ---------------------------------------------------------------------------
__global__ __launch_bounds__(128) void gps_attn_alpha(const float* __restrict__ q,
                                                      const float* __restrict__ k,
                                                      const float* __restrict__ ee,
                                                      const int* __restrict__ EI,
                                                      float* __restrict__ alpha,
                                                      int* __restrict__ mkey, int E)
{
    int e = blockIdx.x;
    int h = threadIdx.x >> 5, lane = threadIdx.x & 31;
    int src = EI[e], dst = EI[E + e];
    int c = h * 32 + lane;
    float p = q[(size_t)dst * 128 + c] * (k[(size_t)src * 128 + c] + ee[(size_t)e * 128 + c]);
    #pragma unroll
    for (int o = 16; o > 0; o >>= 1) p += __shfl_xor(p, o);
    if (lane == 0) {
        float al = p * 0.17677669529663687f; // 1/sqrt(32)
        alpha[e * 4 + h] = al;
        atomicMax(&mkey[dst * 4 + h], fenc(al));
    }
}

__global__ __launch_bounds__(256) void gps_attn_exp(const float* __restrict__ alpha,
                                                    const int* __restrict__ mkey,
                                                    const int* __restrict__ EI,
                                                    float* __restrict__ exb,
                                                    float* __restrict__ den, int E)
{
    int i = blockIdx.x * 256 + threadIdx.x;
    if (i >= E * 4) return;
    int e = i >> 2, h = i & 3;
    int dst = EI[E + e];
    int kk = mkey[dst * 4 + h];
    float m = (kk == MKEY_SENTINEL) ? 0.f : fdec(kk);
    float ex = expf(alpha[i] - m);
    exb[i] = ex;
    atomicAdd(&den[dst * 4 + h], ex);
}

__global__ __launch_bounds__(128) void gps_attn_out(const float* __restrict__ vv,
                                                    const float* __restrict__ ee,
                                                    const float* __restrict__ exb,
                                                    const float* __restrict__ den,
                                                    const int* __restrict__ EI,
                                                    float* __restrict__ outv, int E)
{
    int e = blockIdx.x;
    int c = threadIdx.x;
    int h = c >> 5;
    int src = EI[e], dst = EI[E + e];
    float w = exb[e * 4 + h] / (den[dst * 4 + h] + 1e-16f);
    atomicAdd(&outv[(size_t)dst * 128 + c],
              w * (vv[(size_t)src * 128 + c] + ee[(size_t)e * 128 + c]));
}

// ---------------------------------------------------------------------------
// hEV = [hV[src], hE, hV[dst]]  (E x 384)
// ---------------------------------------------------------------------------
__global__ __launch_bounds__(384) void gps_hev(const float* __restrict__ x,
                                               const float* __restrict__ e,
                                               const int* __restrict__ EI,
                                               float* __restrict__ hEV, int E)
{
    int eid = blockIdx.x;
    int j = threadIdx.x;
    int src = EI[eid], dst = EI[E + eid];
    float val;
    if (j < 128)       val = x[(size_t)src * 128 + j];
    else if (j < 256)  val = e[(size_t)eid * 128 + (j - 128)];
    else               val = x[(size_t)dst * 128 + (j - 256)];
    hEV[(size_t)eid * 384 + j] = val;
}

// ---------------------------------------------------------------------------
// graph pooling, gate MLP, gate apply
// ---------------------------------------------------------------------------
__global__ __launch_bounds__(256) void gps_pool(const float* __restrict__ x,
                                                const int* __restrict__ batch,
                                                float* __restrict__ cV,
                                                float* __restrict__ cnt, int N)
{
    int i = blockIdx.x * 256 + threadIdx.x;
    if (i >= N * 128) return;
    int n = i >> 7, c = i & 127;
    int b = batch[n];
    atomicAdd(&cV[b * 128 + c], x[i]);
    if (c == 0) atomicAdd(&cnt[b], 1.0f);
}

__global__ __launch_bounds__(128) void gps_gate(const float* __restrict__ cV,
                                                const float* __restrict__ cnt,
                                                const float* __restrict__ w1,
                                                const float* __restrict__ b1,
                                                const float* __restrict__ w2,
                                                const float* __restrict__ b2,
                                                float* __restrict__ gate)
{
    __shared__ float cv[128], t1[128];
    int g = blockIdx.x, c = threadIdx.x;
    cv[c] = cV[g * 128 + c] / fmaxf(cnt[g], 1.0f);
    __syncthreads();
    float s = b1[c];
    for (int k = 0; k < 128; k++) s += w1[c * 128 + k] * cv[k];
    t1[c] = fmaxf(s, 0.f);
    __syncthreads();
    float s2 = b2[c];
    for (int k = 0; k < 128; k++) s2 += w2[c * 128 + k] * t1[k];
    gate[g * 128 + c] = 1.f / (1.f + expf(-s2));
}

__global__ __launch_bounds__(256) void gps_gate_apply(float* __restrict__ x,
                                                      const int* __restrict__ batch,
                                                      const float* __restrict__ gate, int N)
{
    int i = blockIdx.x * 256 + threadIdx.x;
    if (i >= N * 128) return;
    int n = i >> 7, c = i & 127;
    x[i] *= gate[batch[n] * 128 + c];
}

// ---------------------------------------------------------------------------
// projection 128 -> 1
// ---------------------------------------------------------------------------
__global__ __launch_bounds__(256) void gps_proj(const float* __restrict__ x,
                                                const float* __restrict__ w,
                                                const float* __restrict__ b,
                                                float* __restrict__ out, int N)
{
    int n = blockIdx.x * 256 + threadIdx.x;
    if (n >= N) return;
    float s = b[0];
    #pragma unroll 4
    for (int k = 0; k < 128; k++) s += w[k] * x[(size_t)n * 128 + k];
    out[n] = s;
}

// ---------------------------------------------------------------------------
// host-side orchestration
// ---------------------------------------------------------------------------
extern "C" void kernel_launch(void* const* d_in, const int* in_sizes, int n_in,
                              void* d_out, int out_size, void* d_ws, size_t ws_size,
                              hipStream_t stream)
{
    (void)in_sizes; (void)n_in; (void)out_size; (void)ws_size;
    const int N = 10000, E = 300000;

    const float* coord = (const float*)d_in[0];
    const float* hVin  = (const float*)d_in[1];
    // params: JAX pytree order (sorted keys, nested)
    const float* bn_ee_b = (const float*)d_in[2];
    const float* bn_ee_g = (const float*)d_in[3];
    const float* bn_n_b  = (const float*)d_in[4];
    const float* bn_n_g  = (const float*)d_in[5];
    const float* eemb_b  = (const float*)d_in[6];
    const float* eemb_w  = (const float*)d_in[7];
    const float* k1_b = (const float*)d_in[8],  *k1_w = (const float*)d_in[9];
    const float* k2_b = (const float*)d_in[10], *k2_w = (const float*)d_in[11];
    const float* k3_b = (const float*)d_in[12], *k3_w = (const float*)d_in[13];
    const int LBASE = 14, LSTRIDE = 26;
    const float* nemb_b = (const float*)d_in[118];
    const float* nemb_w = (const float*)d_in[119];
    const float* proj_b = (const float*)d_in[120];
    const float* proj_w = (const float*)d_in[121];
    const float* we_b = (const float*)d_in[122], *we_w = (const float*)d_in[123];
    const float* wv_b = (const float*)d_in[124], *wv_w = (const float*)d_in[125];
    const int* EI    = (const int*)d_in[126];
    const int* batch = (const int*)d_in[127];
    float* out = (float*)d_out;

    enum { AE_B=0, AE_W, AK_B, AK_W, AQ_B, AQ_W, AV_B, AV_W,
           BNE_B, BNE_G, F1_B, F1_W, F2_B, F2_W,
           G1_B, G1_W, G2_B, G2_W, LN1_B, LN1_G,
           LN2_B, LN2_G, W11_B, W11_W, W12_B, W12_W };
    auto LP = [&](int l, int k) { return (const float*)d_in[LBASE + l * LSTRIDE + k]; };

    // ---- workspace layout (256B aligned, explicit reuse) ----
    char* ws = (char*)d_ws;
    size_t off = 0;
    auto alloc = [&](size_t bytes) -> float* {
        float* p = (float*)(ws + off);
        off += (bytes + 255) & ~(size_t)255;
        return p;
    };
    float* bigA = alloc((size_t)E * EIN_LDA * 4); // x_in (N x 1248) -> e_in (E x 480) -> hEV (E x 384)
    float* B1   = alloc((size_t)E * 128 * 4);     // ee / bn-ed edge emb / gelu(w11)
    float* B2   = alloc((size_t)E * 128 * 4);     // raw edge emb / msg
    float* eP   = alloc((size_t)E * 128 * 4);     // persistent edge features
    float* lc   = alloc((size_t)N * 9 * 4);
    float* xP   = alloc((size_t)N * 128 * 4);     // persistent node features
    float* nt0  = alloc((size_t)N * 128 * 4);
    float* nt1  = alloc((size_t)N * 128 * 4);
    float* nt2  = alloc((size_t)N * 128 * 4);
    float* nt3  = alloc((size_t)N * 128 * 4);
    float* nt4  = alloc((size_t)N * 512 * 4);
    float* alphab = alloc((size_t)E * 4 * 4);
    float* exb    = alloc((size_t)E * 4 * 4);
    int*   mkey   = (int*)alloc((size_t)N * 4 * 4);
    float* den    = alloc((size_t)N * 4 * 4);
    float* stats  = alloc(256 * 4);
    float* cnt    = alloc(32 * 4);
    float* cV     = alloc(32 * 128 * 4);
    float* gate   = alloc(32 * 128 * 4);

    auto fillf = [&](float* p, float v, long long n) {
        gps_fill_f32<<<(unsigned)((n + 255) / 256), 256, 0, stream>>>(p, v, n);
    };
    auto filli = [&](int* p, int v, long long n) {
        gps_fill_i32<<<(unsigned)((n + 255) / 256), 256, 0, stream>>>(p, v, n);
    };
    // pack an f32 weight (M,K) to bf16 (M,Kpad)
    auto pack = [&](const float* W, int M, int K) -> bf16* {
        int Kp = (K + 31) & ~31;
        bf16* p = (bf16*)alloc((size_t)M * Kp * 2);
        long long tot = (long long)M * Kp;
        gps_pack_w<<<(unsigned)((tot + 255) / 256), 256, 0, stream>>>(W, p, M, K, Kp);
        return p;
    };
    auto gemm = [&](const float* X, const bf16* Wp, const float* b, float* Y,
                    int n, int kpad, int m, int act) {
        dim3 grid((n + 127) / 128, m / 64);
        gps_gemm<<<grid, 256, 0, stream>>>(X, Wp, b, Y, n, kpad, m, act);
    };
    auto bn = [&](const float* A, const float* Badd, const float* g, const float* b,
                  float* Y, long long rows) {
        long long total = rows * 128;
        fillf(stats, 0.f, 256);
        int nblk = (int)((total + 255) / 256);
        if (nblk > 4096) nblk = 4096;
        gps_bn_stats<<<nblk, 256, 0, stream>>>(A, Badd, stats, total);
        gps_bn_apply<<<(unsigned)((total + 255) / 256), 256, 0, stream>>>(
            A, Badd, stats, g, b, Y, total, 1.0f / (float)rows);
    };

    // ---- pack all GEMM weights to bf16 (zero-padded K) ----
    bf16* Pnemb = pack(nemb_w, 128, 1217);
    bf16* Peemb = pack(eemb_w, 128, 450);
    bf16* Pwv   = pack(wv_w, 128, 128);
    bf16* Pwe   = pack(we_w, 128, 128);
    bf16* Pk1   = pack(k1_w, 128, 128);
    bf16* Pk2   = pack(k2_w, 128, 128);
    bf16* Pk3   = pack(k3_w, 128, 128);
    bf16 *Pq[4], *Pk[4], *Pv[4], *Pe[4], *Pf1[4], *Pf2[4], *Pw11[4], *Pw12[4];
    for (int l = 0; l < 4; l++) {
        Pq[l]   = pack(LP(l, AQ_W), 128, 128);
        Pk[l]   = pack(LP(l, AK_W), 128, 128);
        Pv[l]   = pack(LP(l, AV_W), 128, 128);
        Pe[l]   = pack(LP(l, AE_W), 128, 128);
        Pf1[l]  = pack(LP(l, F1_W), 512, 128);
        Pf2[l]  = pack(LP(l, F2_W), 128, 512);
        Pw11[l] = pack(LP(l, W11_W), 128, 384);
        Pw12[l] = pack(LP(l, W12_W), 128, 128);
    }

    // ================= node embedding pipeline =================
    gps_copy_hv<<<N, 256, 0, stream>>>(hVin, bigA, N);
    gps_geo_node<<<(N + 127) / 128, 128, 0, stream>>>(coord, bigA, lc, N);
    gps_pad_zero<<<(N * (XIN_LDA - 1217) + 255) / 256, 256, 0, stream>>>(bigA, N, 1217, XIN_LDA);
    gemm(bigA, Pnemb, nemb_b, nt4, N, XIN_LDA, 128, 0);
    bn(nt4, nullptr, bn_n_g, bn_n_b, nt0, N);
    gemm(nt0, Pwv, wv_b, xP, N, 128, 128, 0);

    // ================= edge embedding pipeline =================
    gps_geo_edge<<<(E + 127) / 128, 128, 0, stream>>>(coord, lc, EI, bigA, E);
    gps_pad_zero<<<((long long)E * (EIN_LDA - 450) + 255) / 256, 256, 0, stream>>>(bigA, E, 450, EIN_LDA);
    gemm(bigA, Peemb, eemb_b, B2, E, EIN_LDA, 128, 0);
    bn(B2, nullptr, bn_ee_g, bn_ee_b, B1, E);
    gemm(B1, Pwe, we_b, eP, E, 128, 128, 0);

    // ================= 4 GNN layers =================
    for (int l = 0; l < 4; l++) {
        // --- transformer conv ---
        gemm(xP, Pq[l], LP(l, AQ_B), nt0, N, 128, 128, 0); // q
        gemm(xP, Pk[l], LP(l, AK_B), nt1, N, 128, 128, 0); // k
        gemm(xP, Pv[l], LP(l, AV_B), nt2, N, 128, 128, 0); // v
        gemm(eP, Pe[l], LP(l, AE_B), B1, E, 128, 128, 0);  // ee

        filli(mkey, MKEY_SENTINEL, (long long)N * 4);
        gps_attn_alpha<<<E, 128, 0, stream>>>(nt0, nt1, B1, EI, alphab, mkey, E);
        fillf(den, 0.f, (long long)N * 4);
        gps_attn_exp<<<(E * 4 + 255) / 256, 256, 0, stream>>>(alphab, mkey, EI, exb, den, E);
        fillf(nt3, 0.f, (long long)N * 128);
        gps_attn_out<<<E, 128, 0, stream>>>(nt2, B1, exb, den, EI, nt3, E);

        // hV = LN1(hV + attn)
        gps_ln<<<(N + 7) / 8, 256, 0, stream>>>(xP, nt3, LP(l, LN1_G), LP(l, LN1_B), xP, N);

        // FFN + LN2
        gemm(xP, Pf1[l], LP(l, F1_B), nt4, N, 128, 512, 1);  // relu
        gemm(nt4, Pf2[l], LP(l, F2_B), nt0, N, 512, 128, 0);
        gps_ln<<<(N + 7) / 8, 256, 0, stream>>>(xP, nt0, LP(l, LN2_G), LP(l, LN2_B), xP, N);

        // edge update: msg = w12(gelu(w11([hV_src, hE, hV_dst]))); hE = BN(hE + msg)
        gps_hev<<<E, 384, 0, stream>>>(xP, eP, EI, bigA, E);
        gemm(bigA, Pw11[l], LP(l, W11_B), B1, E, 384, 128, 2); // gelu
        gemm(B1, Pw12[l], LP(l, W12_B), B2, E, 128, 128, 0);
        bn(eP, B2, LP(l, BNE_G), LP(l, BNE_B), eP, E);

        // context gate
        fillf(cnt, 0.f, 32);
        fillf(cV, 0.f, 32 * 128);
        gps_pool<<<(N * 128 + 255) / 256, 256, 0, stream>>>(xP, batch, cV, cnt, N);
        gps_gate<<<32, 128, 0, stream>>>(cV, cnt, LP(l, G1_W), LP(l, G1_B),
                                         LP(l, G2_W), LP(l, G2_B), gate);
        gps_gate_apply<<<(N * 128 + 255) / 256, 256, 0, stream>>>(xP, batch, gate, N);
    }

    // ================= head =================
    gemm(xP, Pk1, k1_b, nt0, N, 128, 128, 1);
    gemm(nt0, Pk2, k2_b, nt1, N, 128, 128, 1);
    gemm(nt1, Pk3, k3_b, nt0, N, 128, 128, 1);
    gps_proj<<<(N + 255) / 256, 256, 0, stream>>>(nt0, proj_w, proj_b, out, N);
}
